// QwenMoEFFN_16879221473406
// MI455X (gfx1250) — compile-verified, matrix operand order checked by reading, not run on hardware
//
#include <hip/hip_runtime.h>
#include <stdint.h>

// ---------------------------------------------------------------------------
// QwenMoE FFN (top-2, capacity 1280) for MI455X / gfx1250, wave32 + WMMA bf16.
//
// Sparse routing (identical math to dense reference; dropped pairs weigh 0):
//   router -> top2 -> capacity keep-first (flat order) -> per-expert gather
//   -> [Me,2048]x[2048,1408] gate/up GEMMs (bf16 WMMA, f32 acc) -> SiLU*up
//   -> [Me,1408]x[1408,2048] down GEMM -> weighted atomic scatter into y.
//
// fp32->bf16 conversion is hoisted out of the GEMM loops (weights + gathered
// activations converted once; bf16 weights fit the 192MB L2 so m-group
// re-reads stay on-chip). GEMM inner loops are straight-line: uniform-base +
// u32-offset b128 loads feeding v_wmma_f32_16x16x32_bf16, with all 8 M-tiles
// computed unconditionally (reads stay inside allocated pad rows; only the
// stores are guarded) so the scheduler can pipeline loads across WMMAs.
// ---------------------------------------------------------------------------

#define E_    8
#define N_    4096          // B*T
#define H_    2048
#define I_    1408
#define CAP_  1280          // int(1.25 * (2N/E))

typedef __attribute__((ext_vector_type(16))) __bf16 v16bf;
typedef __attribute__((ext_vector_type(2)))  __bf16 v2bf;
typedef __attribute__((ext_vector_type(8)))  float  v8f;

union FragBF {
  v16bf v;
  unsigned int w[8];   // two bf16 per word; word j holds elements 2j (lo), 2j+1 (hi)
};

__device__ __forceinline__ unsigned int pk2bf(float lo, float hi) {
#if __has_builtin(__builtin_amdgcn_cvt_pk_bf16_f32)
  v2bf r = __builtin_amdgcn_cvt_pk_bf16_f32(lo, hi);
  return __builtin_bit_cast(unsigned int, r);
#else
  unsigned int a = __float_as_uint(lo);
  unsigned int b = __float_as_uint(hi);
  a = (a + 0x7FFFu + ((a >> 16) & 1u)) >> 16;
  b = (b + 0x7FFFu + ((b >> 16) & 1u)) & 0xFFFF0000u;
  return a | b;
#endif
}

__device__ __forceinline__ unsigned short f2bf(float f) {
  unsigned int x = __float_as_uint(f);
  return (unsigned short)((x + 0x7FFFu + ((x >> 16) & 1u)) >> 16);
}

// 16 bf16 fragment from uniform base + per-lane byte offset (2 b128 loads;
// the +32 for the second chunk folds into the instruction immediate).
__device__ __forceinline__ void loadFragH(FragBF& f, const char* __restrict__ base,
                                          unsigned int off) {
  uint4 t0 = *(const uint4*)(base + off);
  uint4 t1 = *(const uint4*)(base + off + 32);
  f.w[0] = t0.x; f.w[1] = t0.y; f.w[2] = t0.z; f.w[3] = t0.w;
  f.w[4] = t1.x; f.w[5] = t1.y; f.w[6] = t1.z; f.w[7] = t1.w;
}

// 16 fp32 -> bf16 fragment (fallback path, conversion in loop)
__device__ __forceinline__ void loadFragF(FragBF& f, const float* __restrict__ p) {
  const float4* q = (const float4*)p;
  float4 a = q[0], b = q[1], c = q[4], d = q[5];   // elements 0..7 and 16..23
  f.w[0] = pk2bf(a.x, a.y);
  f.w[1] = pk2bf(a.z, a.w);
  f.w[2] = pk2bf(b.x, b.y);
  f.w[3] = pk2bf(b.z, b.w);
  f.w[4] = pk2bf(c.x, c.y);
  f.w[5] = pk2bf(c.z, c.w);
  f.w[6] = pk2bf(d.x, d.y);
  f.w[7] = pk2bf(d.z, d.w);
}

__device__ __forceinline__ uint4 cvt8(const float* __restrict__ src) {
  const float4* q = (const float4*)src;
  float4 a = q[0], b = q[1];
  uint4 o;
  o.x = pk2bf(a.x, a.y);
  o.y = pk2bf(a.z, a.w);
  o.z = pk2bf(b.x, b.y);
  o.w = pk2bf(b.z, b.w);
  return o;
}

__device__ __forceinline__ v8f vzero8() {
  v8f z;
#pragma unroll
  for (int i = 0; i < 8; ++i) z[i] = 0.0f;
  return z;
}

// ---------------------------------------------------------------------------
__global__ void zero_out_kernel(float4* __restrict__ out, int n4) {
  int i = blockIdx.x * blockDim.x + threadIdx.x;
  if (i < n4) out[i] = make_float4(0.f, 0.f, 0.f, 0.f);
}

// fp32 -> bf16 bulk weight conversion; one uint4 (8 elements) per thread.
__global__ void convert_w_kernel(const float* __restrict__ src,
                                 unsigned short* __restrict__ dst, int n8) {
  int i = blockIdx.x * blockDim.x + threadIdx.x;
  if (i >= n8) return;
  ((uint4*)dst)[i] = cvt8(src + (size_t)i * 8);
}

// One block per token; wave e computes logit_e; thread 0 does softmax + top2.
__global__ void router_kernel(const float* __restrict__ x, const float* __restrict__ Wr,
                              int* __restrict__ flat_e, float* __restrict__ flat_val) {
  __shared__ float slog[E_];
  int t = blockIdx.x;
  int wave = threadIdx.x >> 5, lane = threadIdx.x & 31;
  const float* xr = x + (size_t)t * H_;
  const float* wr = Wr + (size_t)wave * H_;
  float s = 0.f;
  for (int k = lane; k < H_; k += 32) s += xr[k] * wr[k];
#pragma unroll
  for (int off = 16; off > 0; off >>= 1) s += __shfl_xor(s, off, 32);
  if (lane == 0) slog[wave] = s;
  __syncthreads();
  if (threadIdx.x == 0) {
    float mx = slog[0];
    for (int e = 1; e < E_; ++e) mx = fmaxf(mx, slog[e]);
    float p[E_], den = 0.f;
    for (int e = 0; e < E_; ++e) { p[e] = __expf(slog[e] - mx); den += p[e]; }
    float inv = 1.f / den;
    for (int e = 0; e < E_; ++e) p[e] *= inv;
    int i0 = 0;
    for (int e = 1; e < E_; ++e) if (p[e] > p[i0]) i0 = e;     // first max on ties
    int i1 = (i0 == 0) ? 1 : 0;
    for (int e = 0; e < E_; ++e) { if (e == i0) continue; if (p[e] > p[i1]) i1 = e; }
    flat_e[t]      = i0; flat_val[t]      = p[i0];   // slot 0, flat order [0..N)
    flat_e[N_ + t] = i1; flat_val[N_ + t] = p[i1];   // slot 1, flat order [N..2N)
  }
}

// One wave per expert scans all 2N assignments in reference flat order,
// using ballot prefix-counts to compute the exclusive rank; keep rank < cap.
__global__ void assign_kernel(const int* __restrict__ flat_e, const float* __restrict__ flat_val,
                              int* __restrict__ counts, int* __restrict__ etok,
                              float* __restrict__ ewt) {
  int e = threadIdx.x >> 5, lane = threadIdx.x & 31;
  int running = 0;
  for (int base = 0; base < 2 * N_; base += 32) {
    int s = base + lane;
    bool match = (flat_e[s] == e);
    unsigned int m = (unsigned int)__ballot(match);
    int before = __popc(m & ((1u << lane) - 1u));
    int rank = running + before;                    // 0-based rank within expert
    if (match && rank < CAP_) {
      etok[e * CAP_ + rank] = (s < N_) ? s : (s - N_);
      ewt [e * CAP_ + rank] = flat_val[s];
    }
    running += __popc(m);
  }
  if (lane == 0) counts[e] = (running < CAP_) ? running : CAP_;
}

// Gather routed token rows and convert to bf16: abuf[e][m][0..H) (m < M_pad).
// One block per (expert, slot); 256 threads x 8 elements = H.
__global__ void gather_x_kernel(const float* __restrict__ x, const int* __restrict__ counts,
                                const int* __restrict__ etok,
                                unsigned short* __restrict__ abuf) {
  int e = blockIdx.x / CAP_;
  int m = blockIdx.x % CAP_;
  int count = counts[e];
  if (count <= 0) return;
  int M_pad = (count + 15) & ~15;
  if (m >= M_pad) return;
  int mc = (m < count) ? m : (count - 1);           // clamp pad rows (finite data)
  const float* src = x + (size_t)etok[e * CAP_ + mc] * H_;
  unsigned short* dst = abuf + ((size_t)e * CAP_ + m) * H_;
  int i = threadIdx.x;                              // H_/8 == 256 exactly
  ((uint4*)dst)[i] = cvt8(src + (size_t)i * 8);
}

// ---------------------------------------------------------------------------
// Fused gate/up projection + SiLU*up, bf16 WMMA, writes bf16 hbuf[e][m][i].
// Wave: one 16-wide I column tile, 8 M-tiles (128 rows), K=H=2048 in steps of 32.
// All 8 M-tiles are computed unconditionally (reads stay within the CAP_ pad
// rows); only the hbuf stores are guarded by M_pad.
template <bool PRE>
__global__ void __launch_bounds__(256)
gateup_kernel(const float* __restrict__ x, const unsigned short* __restrict__ abuf,
              const float* __restrict__ Wg, const float* __restrict__ Wu,
              const unsigned short* __restrict__ Wgh, const unsigned short* __restrict__ Wuh,
              const int* __restrict__ counts, const int* __restrict__ etok,
              unsigned short* __restrict__ hbuf) {
  int e = blockIdx.z;
  int count = counts[e];
  if (count <= 0) return;
  int M_pad = (count + 15) & ~15;
  int wave = threadIdx.x >> 5, lane = threadIdx.x & 31;
  int n_col = (blockIdx.x * 8 + wave) * 16 + (lane & 15);   // output channel in [0,1408)
  int m_base = blockIdx.y * 128;
  if (m_base >= M_pad) return;
  int half  = (lane < 16) ? 0 : 8;   // ISA 16-bit A/B fragment K-offset per lane half
  int mlane = lane & 15;

  v8f accg[8], accu[8];
#pragma unroll
  for (int mt = 0; mt < 8; ++mt) { accg[mt] = vzero8(); accu[mt] = vzero8(); }

  if constexpr (PRE) {
    // Uniform SGPR bases + 32-bit per-lane byte offsets -> saddr-form loads.
    const char* abase = (const char*)(abuf + (size_t)e * CAP_ * H_);
    const char* gbase = (const char*)(Wgh  + (size_t)e * I_ * H_);
    const char* ubase = (const char*)(Wuh  + (size_t)e * I_ * H_);
    unsigned int aoff[8];
#pragma unroll
    for (int mt = 0; mt < 8; ++mt)
      aoff[mt] = (unsigned int)(((m_base + mt * 16 + mlane) * H_ + half) * 2);
    unsigned int goff = (unsigned int)((n_col * H_ + half) * 2);
    unsigned int uoff = goff;

    for (int k = 0; k < H_; k += 32) {
      FragBF bg, bu;
      loadFragH(bg, gbase, goff);
      loadFragH(bu, ubase, uoff);
#pragma unroll
      for (int mt = 0; mt < 8; ++mt) {
        FragBF a;
        loadFragH(a, abase, aoff[mt]);
        accg[mt] = __builtin_amdgcn_wmma_f32_16x16x32_bf16(
            false, a.v, false, bg.v, (short)0, accg[mt], false, false);
        accu[mt] = __builtin_amdgcn_wmma_f32_16x16x32_bf16(
            false, a.v, false, bu.v, (short)0, accu[mt], false, false);
        aoff[mt] += 64;
      }
      goff += 64;
      uoff += 64;
    }
  } else {
    const float* xrows[8];
#pragma unroll
    for (int mt = 0; mt < 8; ++mt) {
      int m = m_base + mt * 16 + mlane;
      int mc = (m < count) ? m : (count - 1);       // clamp pad rows (finite data)
      xrows[mt] = x + (size_t)etok[e * CAP_ + mc] * H_;
    }
    const float* bgp = Wg + ((size_t)e * I_ + n_col) * H_;
    const float* bup = Wu + ((size_t)e * I_ + n_col) * H_;
    for (int k = 0; k < H_; k += 32) {
      FragBF bg, bu;
      loadFragF(bg, bgp + k + half);
      loadFragF(bu, bup + k + half);
#pragma unroll
      for (int mt = 0; mt < 8; ++mt) {
        FragBF a;
        loadFragF(a, xrows[mt] + k + half);
        accg[mt] = __builtin_amdgcn_wmma_f32_16x16x32_bf16(
            false, a.v, false, bg.v, (short)0, accg[mt], false, false);
        accu[mt] = __builtin_amdgcn_wmma_f32_16x16x32_bf16(
            false, a.v, false, bu.v, (short)0, accu[mt], false, false);
      }
    }
  }

  // SiLU(gate) * up -> bf16 hbuf. D layout: lane l: N=l%16; vgpr r: M=r+(l<16?0:8).
  unsigned short* hb = hbuf + (size_t)e * CAP_ * I_;
#pragma unroll
  for (int mt = 0; mt < 8; ++mt) {
    if (m_base + mt * 16 >= M_pad) continue;
#pragma unroll
    for (int r = 0; r < 8; ++r) {
      int m = m_base + mt * 16 + r + half;
      float g = accg[mt][r];
      float u = accu[mt][r];
      float hv = (g / (1.f + __expf(-g))) * u;
      hb[(size_t)m * I_ + n_col] = f2bf(hv);
    }
  }
}

// Down projection + weighted scatter-add: y[token] += w * (h row · Wd col).
template <bool PRE>
__global__ void __launch_bounds__(256)
down_kernel(const unsigned short* __restrict__ hbuf,
            const float* __restrict__ Wd, const unsigned short* __restrict__ Wdh,
            const int* __restrict__ counts, const int* __restrict__ etok,
            const float* __restrict__ ewt, float* __restrict__ out) {
  int e = blockIdx.z;
  int count = counts[e];
  if (count <= 0) return;
  int M_pad = (count + 15) & ~15;
  int wave = threadIdx.x >> 5, lane = threadIdx.x & 31;
  int n_col = (blockIdx.x * 8 + wave) * 16 + (lane & 15);   // output channel in [0,2048)
  int m_base = blockIdx.y * 128;
  if (m_base >= M_pad) return;
  int half  = (lane < 16) ? 0 : 8;
  int mlane = lane & 15;

  v8f acc[8];
#pragma unroll
  for (int mt = 0; mt < 8; ++mt) acc[mt] = vzero8();

  const char* hbase = (const char*)(hbuf + (size_t)e * CAP_ * I_);
  unsigned int hoff[8];
#pragma unroll
  for (int mt = 0; mt < 8; ++mt)
    hoff[mt] = (unsigned int)(((m_base + mt * 16 + mlane) * I_ + half) * 2);

  if constexpr (PRE) {
    const char* wbase = (const char*)(Wdh + (size_t)e * H_ * I_);
    unsigned int woff = (unsigned int)((n_col * I_ + half) * 2);
    for (int k = 0; k < I_; k += 32) {
      FragBF bw;
      loadFragH(bw, wbase, woff);
#pragma unroll
      for (int mt = 0; mt < 8; ++mt) {
        FragBF a;
        loadFragH(a, hbase, hoff[mt]);
        acc[mt] = __builtin_amdgcn_wmma_f32_16x16x32_bf16(
            false, a.v, false, bw.v, (short)0, acc[mt], false, false);
        hoff[mt] += 64;
      }
      woff += 64;
    }
  } else {
    const float* bwp = Wd + ((size_t)e * H_ + n_col) * I_;
    for (int k = 0; k < I_; k += 32) {
      FragBF bw;
      loadFragF(bw, bwp + k + half);
#pragma unroll
      for (int mt = 0; mt < 8; ++mt) {
        FragBF a;
        loadFragH(a, hbase, hoff[mt]);
        acc[mt] = __builtin_amdgcn_wmma_f32_16x16x32_bf16(
            false, a.v, false, bw.v, (short)0, acc[mt], false, false);
        hoff[mt] += 64;
      }
    }
  }

#pragma unroll
  for (int mt = 0; mt < 8; ++mt) {
    if (m_base + mt * 16 >= M_pad) continue;
#pragma unroll
    for (int r = 0; r < 8; ++r) {
      int m = m_base + mt * 16 + r + half;
      if (m < count) {
        int token = etok[e * CAP_ + m];
        float w = ewt[e * CAP_ + m];
        atomicAdd(out + (size_t)token * H_ + n_col, w * acc[mt][r]);
      }
    }
  }
}

// ---------------------------------------------------------------------------
extern "C" void kernel_launch(void* const* d_in, const int* in_sizes, int n_in,
                              void* d_out, int out_size, void* d_ws, size_t ws_size,
                              hipStream_t stream) {
  (void)in_sizes; (void)n_in; (void)out_size;
  const float* x  = (const float*)d_in[0];
  const float* Wr = (const float*)d_in[1];
  const float* Wg = (const float*)d_in[2];
  const float* Wu = (const float*)d_in[3];
  const float* Wd = (const float*)d_in[4];
  float* out = (float*)d_out;

  char* ws = (char*)d_ws;
  size_t off = 0;
  auto alloc = [&](size_t bytes) -> char* {
    char* p = ws + off;
    off = (off + bytes + 255) & ~(size_t)255;
    return p;
  };
  // --- common workspace (~29 MB) ---
  int*   flat_e   = (int*)  alloc((size_t)2 * N_ * sizeof(int));
  float* flat_val = (float*)alloc((size_t)2 * N_ * sizeof(float));
  int*   counts   = (int*)  alloc((size_t)E_ * sizeof(int));
  int*   etok     = (int*)  alloc((size_t)E_ * CAP_ * sizeof(int));
  float* ewt      = (float*)alloc((size_t)E_ * CAP_ * sizeof(float));
  unsigned short* hbuf = (unsigned short*)alloc((size_t)E_ * CAP_ * I_ * sizeof(unsigned short));
  // --- extended workspace for preconverted bf16 operands (~180 MB more) ---
  unsigned short* abuf = (unsigned short*)alloc((size_t)E_ * CAP_ * H_ * sizeof(unsigned short));
  unsigned short* Wgh  = (unsigned short*)alloc((size_t)E_ * I_ * H_ * sizeof(unsigned short));
  unsigned short* Wuh  = (unsigned short*)alloc((size_t)E_ * I_ * H_ * sizeof(unsigned short));
  unsigned short* Wdh  = (unsigned short*)alloc((size_t)E_ * H_ * I_ * sizeof(unsigned short));
  const bool pre = (ws_size >= off);          // fixed per process -> deterministic

  zero_out_kernel<<<(N_ * H_ / 4 + 255) / 256, 256, 0, stream>>>((float4*)out, N_ * H_ / 4);
  router_kernel<<<N_, 256, 0, stream>>>(x, Wr, flat_e, flat_val);
  assign_kernel<<<1, 256, 0, stream>>>(flat_e, flat_val, counts, etok, ewt);

  if (pre) {
    const int n8 = E_ * I_ * H_ / 8;          // same element count for all 3 tensors
    convert_w_kernel<<<(n8 + 255) / 256, 256, 0, stream>>>(Wg, Wgh, n8);
    convert_w_kernel<<<(n8 + 255) / 256, 256, 0, stream>>>(Wu, Wuh, n8);
    convert_w_kernel<<<(n8 + 255) / 256, 256, 0, stream>>>(Wd, Wdh, n8);
    gather_x_kernel<<<E_ * CAP_, 256, 0, stream>>>(x, counts, etok, abuf);
    gateup_kernel<true><<<dim3(I_ / 128, CAP_ / 128, E_), 256, 0, stream>>>(
        x, abuf, Wg, Wu, Wgh, Wuh, counts, etok, hbuf);
    down_kernel<true><<<dim3(H_ / 128, CAP_ / 128, E_), 256, 0, stream>>>(
        hbuf, Wd, Wdh, counts, etok, ewt, out);
  } else {
    gateup_kernel<false><<<dim3(I_ / 128, CAP_ / 128, E_), 256, 0, stream>>>(
        x, abuf, Wg, Wu, Wgh, Wuh, counts, etok, hbuf);
    down_kernel<false><<<dim3(H_ / 128, CAP_ / 128, E_), 256, 0, stream>>>(
        hbuf, Wd, Wdh, counts, etok, ewt, out);
  }
}
